// GAT_61967788147083
// MI455X (gfx1250) — matrix operand order repaired and checked by reading, not compiled, not run on hardware
//
#include <hip/hip_runtime.h>

// ---------------------------------------------------------------------------
// Problem constants (from reference)
// ---------------------------------------------------------------------------
constexpr int N_NODES = 10000;
constexpr int N_EDGES = 160000;
constexpr int E_TOT   = N_EDGES + N_NODES;   // with self loops
constexpr int F_IN    = 300;
constexpr int D       = 1024;
constexpr int N_CLS   = 10;
constexpr float NEG_SLOPE = 0.2f;

// Monotonic u32 encoding of float(-inf): bits(-inf)=0xFF800000, sign set -> ~u
constexpr unsigned int NEG_INF_ENC = 0x007FFFFFu;

// ---------------------------------------------------------------------------
// CDNA5 types
// ---------------------------------------------------------------------------
typedef __attribute__((ext_vector_type(16))) __bf16        v16bf;
typedef __attribute__((ext_vector_type(8)))  float         v8f;
typedef __attribute__((ext_vector_type(4)))  unsigned int  v4u;
typedef __attribute__((ext_vector_type(8)))  int           v8i;
typedef __attribute__((ext_vector_type(4)))  int           v4i;

union AFrag { v16bf v; unsigned int u[8]; };
union BFrag { v16bf v; unsigned int u[8]; };

__device__ __forceinline__ unsigned short f32_to_bf16(float f) {
    unsigned int u = __float_as_uint(f);
    unsigned int r = u + 0x7FFFu + ((u >> 16) & 1u);   // round-to-nearest-even
    return (unsigned short)(r >> 16);
}

// Monotonic u32 encoding of float for atomicMax on signed floats
__device__ __forceinline__ unsigned int encF(float f) {
    unsigned int u = __float_as_uint(f);
    return (u & 0x80000000u) ? ~u : (u | 0x80000000u);
}
__device__ __forceinline__ float decF(unsigned int k) {
    unsigned int u = (k & 0x80000000u) ? (k & 0x7FFFFFFFu) : ~k;
    return __uint_as_float(u);
}

// LDS aperture: flat shared address carries the LDS byte offset in [31:0]
__device__ __forceinline__ unsigned int lds_off(const void* p) {
    return (unsigned int)(unsigned long long)p;
}

// ---------------------------------------------------------------------------
// fp32 -> packed bf16 staging (2 elems / thread)
// ---------------------------------------------------------------------------
__global__ void f32_to_bf16_pack(const float* __restrict__ in,
                                 unsigned int* __restrict__ out, int nPairs) {
    int i = blockIdx.x * blockDim.x + threadIdx.x;
    if (i < nPairs) {
        float2 v = ((const float2*)in)[i];
        out[i] = (unsigned int)f32_to_bf16(v.x) |
                 ((unsigned int)f32_to_bf16(v.y) << 16);
    }
}

// ---------------------------------------------------------------------------
// TDM 2D tile load: global (bf16, row stride = strideElems) -> LDS with
// hardware row padding.  D# layout per CDNA5 ISA ch.8.
//   padInterval code: pad after (2<<code) DWORDs;  padAmount code: (code+1) DWORDs
// ---------------------------------------------------------------------------
#define HAVE_TDM __has_builtin(__builtin_amdgcn_tensor_load_to_lds)

#if HAVE_TDM
__device__ __forceinline__ void tdm_load_2d(unsigned int ldsOffset,
                                            const unsigned short* gptr,
                                            unsigned int tensorD0,  // remaining elems, dim0 (OOB->0)
                                            unsigned int tensorD1,  // remaining rows,  dim1 (OOB->0)
                                            unsigned int tileD0, unsigned int tileD1,
                                            unsigned long long strideElems,
                                            unsigned int padInterval,
                                            unsigned int padAmount) {
    unsigned long long ga = (unsigned long long)gptr;
    v4u g0;
    g0.x = 1u;                                   // count=1, user descriptor
    g0.y = ldsOffset;                            // lds_addr
    g0.z = (unsigned int)ga;                     // global_addr[31:0]
    g0.w = (unsigned int)((ga >> 32) & 0x01FFFFFFu) | (2u << 30);  // addr[56:32] | type=2
    v8i g1;
    g1[0] = (int)((1u << 16)                     // data_size = 2 bytes
                | (1u << 20)                     // pad_enable
                | (padInterval << 22)
                | (padAmount << 25));
    g1[1] = (int)((tensorD0 & 0xFFFFu) << 16);                       // dim0[15:0]
    g1[2] = (int)((tensorD0 >> 16) | ((tensorD1 & 0xFFFFu) << 16));  // dim0[31:16] | dim1[15:0]
    g1[3] = (int)((tensorD1 >> 16) | (tileD0 << 16));                // dim1[31:16] | tile0
    g1[4] = (int)(tileD1 & 0xFFFFu);                                 // tile1 | tile2=0
    g1[5] = (int)(unsigned int)(strideElems & 0xFFFFFFFFu);          // stride0[31:0]
    g1[6] = (int)(unsigned int)((strideElems >> 32) & 0xFFFFu);      // stride0[47:32] | stride1=0
    g1[7] = 0;
    v4i g2 = {0, 0, 0, 0};   // 2D: dims 2..4 unused
    v4i g3 = {0, 0, 0, 0};
#if __clang_major__ >= 23
    v8i g4 = {0, 0, 0, 0, 0, 0, 0, 0};
    __builtin_amdgcn_tensor_load_to_lds(g0, g1, g2, g3, g4, 0);
#else
    __builtin_amdgcn_tensor_load_to_lds(g0, g1, g2, g3, 0);
#endif
}
#endif

// ---------------------------------------------------------------------------
// Tiled bf16 WMMA GEMM: C[M,N] = A[M,K] @ B[K,N]   (bf16 in, f32 accum/out)
// Block tile 128x128, BK=32, 8 waves (4x2), wave tile 32x64 (2x4 WMMA frags).
// Double-buffered TDM pipeline: tile s+1 DMA overlaps WMMA on tile s.
// ---------------------------------------------------------------------------
#define BM 128
#define BN 128
#define BK 32
#define LDA 34    // A tile LDS stride (bf16): 16 DWORDs/row + 1 pad DWORD
#define LDB 130   // B tile LDS stride (bf16): 64 DWORDs/row + 1 pad DWORD

__global__ __launch_bounds__(256)
void gemm_bf16_wmma(const unsigned short* __restrict__ A,
                    const unsigned short* __restrict__ B,
                    float* __restrict__ C, int M, int N, int K) {
    __shared__ unsigned short As[2][BM * LDA];
    __shared__ unsigned short Bs[2][BK * LDB];

    const int tid     = threadIdx.x;
    const int lane    = tid & 31;
    const int wave    = tid >> 5;     // 0..7
    const int waveRow = wave >> 1;    // 0..3  -> rows [waveRow*32, +32)
    const int waveCol = wave & 1;     // 0..1  -> cols [waveCol*64, +64)

    const int blockM = blockIdx.y * BM;
    const int blockN = blockIdx.x * BN;
    const int steps  = (K + BK - 1) / BK;

    v8f acc[2][4];
#pragma unroll
    for (int mt = 0; mt < 2; ++mt)
#pragma unroll
        for (int nt = 0; nt < 4; ++nt)
            acc[mt][nt] = (v8f)0.0f;

    const int mBase = lane & 15;            // A fragment: M = lane & 15
    const int kHalf = (lane >> 4) << 3;     // lanes 16-31 hold K half offset 8

#if HAVE_TDM
    auto issue = [&](int s) {
        int k0  = s * BK;
        int buf = s & 1;
        // A tile: BM rows x BK cols, row stride K; zero-fill past M / K edges
        tdm_load_2d(lds_off(&As[buf][0]), A + (size_t)blockM * K + k0,
                    (unsigned)(K - k0), (unsigned)(M - blockM),
                    BK, BM, (unsigned long long)K,
                    /*pad: after 16 DWORDs*/ 3u, /*1 DWORD*/ 0u);
        // B tile: BK rows x BN cols, row stride N; zero-fill past K edge
        tdm_load_2d(lds_off(&Bs[buf][0]), B + (size_t)k0 * N + blockN,
                    (unsigned)(N - blockN), (unsigned)(K - k0),
                    BN, BK, (unsigned long long)N,
                    /*pad: after 64 DWORDs*/ 5u, /*1 DWORD*/ 0u);
    };
    if (wave == 0) issue(0);
#endif

    for (int s = 0; s < steps; ++s) {
        const int buf = s & 1;
#if HAVE_TDM
        if (wave == 0) __builtin_amdgcn_s_wait_tensorcnt(0);
        __syncthreads();                       // tile s ready; tile s-1 consumers done
        if (wave == 0 && s + 1 < steps) issue(s + 1);
#else
        // fallback: manual staged copy (bf16 already in global)
        __syncthreads();
        {
            int k0 = s * BK;
            for (int i = tid; i < BM * BK; i += 256) {
                int r = i >> 5, kk = i & 31;
                int gm = blockM + r, gk = k0 + kk;
                As[buf][r * LDA + kk] =
                    (gm < M && gk < K) ? A[(size_t)gm * K + gk] : (unsigned short)0;
            }
            for (int i = tid; i < BK * BN; i += 256) {
                int r = i >> 7, c = i & 127;
                int gk = k0 + r;
                Bs[buf][r * LDB + c] =
                    (gk < K) ? B[(size_t)gk * N + (blockN + c)] : (unsigned short)0;
            }
        }
        __syncthreads();
#endif

        // ---- B fragments: N = lane & 15, elements = K (same K order as A) ----
        BFrag bfr[4];
#pragma unroll
        for (int nt = 0; nt < 4; ++nt) {
            const unsigned short* src =
                &Bs[buf][waveCol * 64 + nt * 16 + mBase];
#pragma unroll
            for (int j = 0; j < 8; ++j) {
                int kk = ((j >= 4) ? 16 : 0) + kHalf + ((j & 3) << 1);
                bfr[nt].u[j] = (unsigned int)src[kk * LDB] |
                               ((unsigned int)src[(kk + 1) * LDB] << 16);
            }
        }

        // ---- A fragments + WMMA ----
#pragma unroll
        for (int mt = 0; mt < 2; ++mt) {
            AFrag af;
            const unsigned short* arow =
                &As[buf][(waveRow * 32 + mt * 16 + mBase) * LDA];
#pragma unroll
            for (int j = 0; j < 8; ++j) {
                // dword j covers K = {kk, kk+1}
                int kk = ((j >= 4) ? 16 : 0) + kHalf + ((j & 3) << 1);
                af.u[j] = *(const unsigned int*)&arow[kk];
            }
#pragma unroll
            for (int nt = 0; nt < 4; ++nt) {
                acc[mt][nt] = __builtin_amdgcn_wmma_f32_16x16x32_bf16(
                    false, af.v, false, bfr[nt].v,
                    (short)0, acc[mt][nt], false, false);
            }
        }
    }

    // ---- epilogue: C/D layout lanes 0-15 -> M=r, lanes 16-31 -> M=r+8 ----
#pragma unroll
    for (int mt = 0; mt < 2; ++mt)
#pragma unroll
        for (int nt = 0; nt < 4; ++nt) {
            int gn   = blockN + waveCol * 64 + nt * 16 + (lane & 15);
            int mrow = blockM + waveRow * 32 + mt * 16 + ((lane >> 4) << 3);
#pragma unroll
            for (int r = 0; r < 8; ++r) {
                int gm = mrow + r;
                if (gm < M) C[(size_t)gm * N + gn] = acc[mt][nt][r];
            }
        }
}

// ---------------------------------------------------------------------------
// Per-node attention scalars: s[i] = h[i].a_src, d[i] = h[i].a_dst
// one wave32 per node
// ---------------------------------------------------------------------------
__global__ __launch_bounds__(256)
void node_dots(const float* __restrict__ h, const float* __restrict__ a_src,
               const float* __restrict__ a_dst, float* __restrict__ s,
               float* __restrict__ d, int nNodes, int dim) {
    int node = (blockIdx.x * blockDim.x + threadIdx.x) >> 5;
    int lane = threadIdx.x & 31;
    if (node >= nNodes) return;
    const float* hr = h + (size_t)node * dim;
    float ss = 0.0f, dd = 0.0f;
    for (int f = lane; f < dim; f += 32) {
        float v = hr[f];
        ss += v * a_src[f];
        dd += v * a_dst[f];
    }
#pragma unroll
    for (int o = 16; o > 0; o >>= 1) {
        ss += __shfl_xor(ss, o, 32);
        dd += __shfl_xor(dd, o, 32);
    }
    if (lane == 0) { s[node] = ss; d[node] = dd; }
}

// ---------------------------------------------------------------------------
// fills
// ---------------------------------------------------------------------------
__global__ void fill_f32(float* p, float v, int n) {
    int i = blockIdx.x * blockDim.x + threadIdx.x;
    if (i < n) p[i] = v;
}
__global__ void fill_u32(unsigned int* p, unsigned int v, int n) {
    int i = blockIdx.x * blockDim.x + threadIdx.x;
    if (i < n) p[i] = v;
}

// ---------------------------------------------------------------------------
// edge decoding helper: self loops appended after real edges
// ---------------------------------------------------------------------------
__device__ __forceinline__ void edge_sd(const long long* ei, int e,
                                        int& src, int& dst) {
    if (e < N_EDGES) {
        src = (int)ei[e];
        dst = (int)ei[N_EDGES + e];
    } else {
        src = dst = e - N_EDGES;
    }
}

// pass 1: logits + scatter-max over dst
__global__ void edge_logit_max(const long long* __restrict__ ei,
                               const float* __restrict__ s,
                               const float* __restrict__ d,
                               float* __restrict__ eLogit,
                               unsigned int* __restrict__ maxEnc) {
    int e = blockIdx.x * blockDim.x + threadIdx.x;
    if (e >= E_TOT) return;
    int src, dst;
    edge_sd(ei, e, src, dst);
    float x  = s[src] + d[dst];
    float lg = x > 0.0f ? x : NEG_SLOPE * x;
    eLogit[e] = lg;
    atomicMax(&maxEnc[dst], encF(lg));
}

// pass 2: exp(e - max[dst]) + scatter-sum denom
__global__ void edge_exp_sum(const long long* __restrict__ ei,
                             const float* __restrict__ eLogit,
                             const unsigned int* __restrict__ maxEnc,
                             float* __restrict__ eEx,
                             float* __restrict__ denom) {
    int e = blockIdx.x * blockDim.x + threadIdx.x;
    if (e >= E_TOT) return;
    int src, dst;
    edge_sd(ei, e, src, dst);
    float ex = __expf(eLogit[e] - decF(maxEnc[dst]));
    eEx[e] = ex;
    atomicAdd(&denom[dst], ex);
}

// pass 3: out[dst] += alpha * h[src]; one 256-thread block per edge, 4 f32/thread
__global__ __launch_bounds__(256)
void edge_aggregate(const long long* __restrict__ ei,
                    const float* __restrict__ eEx,
                    const float* __restrict__ denom,
                    const float* __restrict__ h,
                    float* __restrict__ out) {
    int e = blockIdx.x;
    int src, dst;
    edge_sd(ei, e, src, dst);
    float alpha = eEx[e] / denom[dst];
    const float4* hs = (const float4*)(h + (size_t)src * D);
    float* od = out + (size_t)dst * D;
    float4 v = hs[threadIdx.x];           // D/4 == 256
    int f = threadIdx.x * 4;
    atomicAdd(&od[f + 0], alpha * v.x);
    atomicAdd(&od[f + 1], alpha * v.y);
    atomicAdd(&od[f + 2], alpha * v.z);
    atomicAdd(&od[f + 3], alpha * v.w);
}

// bias + relu (in place)
__global__ void bias_relu(float* __restrict__ x, const float* __restrict__ b,
                          int n, int dim) {
    int i = blockIdx.x * blockDim.x + threadIdx.x;
    if (i < n) {
        float v = x[i] + b[i % dim];
        x[i] = v > 0.0f ? v : 0.0f;
    }
}

// ---------------------------------------------------------------------------
// classifier: softmax(emb @ fc_w + fc_b) — one wave32 per node
// ---------------------------------------------------------------------------
__global__ __launch_bounds__(256)
void fc_softmax(const float* __restrict__ emb, const float* __restrict__ W,
                const float* __restrict__ b, float* __restrict__ out,
                int nNodes, int dim) {
    int node = (blockIdx.x * blockDim.x + threadIdx.x) >> 5;
    int lane = threadIdx.x & 31;
    if (node >= nNodes) return;
    const float* er = emb + (size_t)node * dim;
    float acc[N_CLS];
#pragma unroll
    for (int c = 0; c < N_CLS; ++c) acc[c] = 0.0f;
    for (int f = lane; f < dim; f += 32) {
        float v = er[f];
        const float* wr = W + (size_t)f * N_CLS;
#pragma unroll
        for (int c = 0; c < N_CLS; ++c) acc[c] += v * wr[c];
    }
#pragma unroll
    for (int c = 0; c < N_CLS; ++c)
#pragma unroll
        for (int o = 16; o > 0; o >>= 1)
            acc[c] += __shfl_xor(acc[c], o, 32);
    if (lane == 0) {
        float m = -3.4e38f;
#pragma unroll
        for (int c = 0; c < N_CLS; ++c) { acc[c] += b[c]; m = fmaxf(m, acc[c]); }
        float sum = 0.0f;
#pragma unroll
        for (int c = 0; c < N_CLS; ++c) { acc[c] = __expf(acc[c] - m); sum += acc[c]; }
        float inv = 1.0f / sum;
        float* o = out + (size_t)node * N_CLS;
#pragma unroll
        for (int c = 0; c < N_CLS; ++c) o[c] = acc[c] * inv;
    }
}

// ---------------------------------------------------------------------------
// launcher
// ---------------------------------------------------------------------------
extern "C" void kernel_launch(void* const* d_in, const int* in_sizes, int n_in,
                              void* d_out, int out_size, void* d_ws, size_t ws_size,
                              hipStream_t stream) {
    const float*     x    = (const float*)d_in[0];
    const long long* ei   = (const long long*)d_in[1];   // int64 [2, N_EDGES]
    const float*     W1   = (const float*)d_in[2];
    const float*     as1  = (const float*)d_in[3];
    const float*     ad1  = (const float*)d_in[4];
    const float*     b1   = (const float*)d_in[5];
    const float*     W2   = (const float*)d_in[6];
    const float*     as2  = (const float*)d_in[7];
    const float*     ad2  = (const float*)d_in[8];
    const float*     b2   = (const float*)d_in[9];
    const float*     fcw  = (const float*)d_in[10];
    const float*     fcb  = (const float*)d_in[11];
    float*           outp = (float*)d_out;

    // ---- workspace layout ----
    size_t off = 0;
    auto alloc = [&](size_t bytes) {
        void* p = (char*)d_ws + off;
        off += (bytes + 255) & ~(size_t)255;
        return p;
    };
    float*          H      = (float*)alloc((size_t)N_NODES * D * sizeof(float));
    float*          OUT    = (float*)alloc((size_t)N_NODES * D * sizeof(float));
    unsigned short* Abf    = (unsigned short*)alloc((size_t)N_NODES * D * 2);  // bf16 A stage
    unsigned short* Bbf    = (unsigned short*)alloc((size_t)D * D * 2);        // bf16 B stage
    float*          sArr   = (float*)alloc(N_NODES * sizeof(float));
    float*          dArr   = (float*)alloc(N_NODES * sizeof(float));
    unsigned int*   maxEnc = (unsigned int*)alloc(N_NODES * sizeof(unsigned int));
    float*          denom  = (float*)alloc(N_NODES * sizeof(float));
    float*          eLogit = (float*)alloc(E_TOT * sizeof(float));
    float*          eEx    = (float*)alloc(E_TOT * sizeof(float));
    (void)ws_size; (void)n_in; (void)in_sizes; (void)out_size;

    const int nodeBlk   = (N_NODES * 32 + 255) / 256;   // wave-per-node kernels
    const int edgeBlk   = (E_TOT + 255) / 256;
    const int nodeElems = N_NODES * D;
    const int elemBlk   = (nodeElems + 255) / 256;
    const int nBlk      = (N_NODES + 255) / 256;

    dim3 gemmGrid(D / BN, (N_NODES + BM - 1) / BM);     // (8, 79)

    auto gat_layer = [&](const float* in, int K, const float* W,
                         const float* asv, const float* adv, const float* bias) {
        // stage A (N_NODES x K) and B (K x D) as bf16
        int aPairs = N_NODES * K / 2;
        int bPairs = K * D / 2;
        f32_to_bf16_pack<<<(aPairs + 255) / 256, 256, 0, stream>>>(
            in, (unsigned int*)Abf, aPairs);
        f32_to_bf16_pack<<<(bPairs + 255) / 256, 256, 0, stream>>>(
            W, (unsigned int*)Bbf, bPairs);
        // h = in @ W   (WMMA + TDM double buffering)
        gemm_bf16_wmma<<<gemmGrid, 256, 0, stream>>>(Abf, Bbf, H, N_NODES, D, K);
        // attention scalars
        node_dots<<<nodeBlk, 256, 0, stream>>>(H, asv, adv, sArr, dArr, N_NODES, D);
        // init scatter buffers
        fill_u32<<<nBlk, 256, 0, stream>>>(maxEnc, NEG_INF_ENC, N_NODES);
        fill_f32<<<nBlk, 256, 0, stream>>>(denom, 0.0f, N_NODES);
        fill_f32<<<elemBlk, 256, 0, stream>>>(OUT, 0.0f, nodeElems);
        // edge softmax + aggregation
        edge_logit_max<<<edgeBlk, 256, 0, stream>>>(ei, sArr, dArr, eLogit, maxEnc);
        edge_exp_sum<<<edgeBlk, 256, 0, stream>>>(ei, eLogit, maxEnc, eEx, denom);
        edge_aggregate<<<E_TOT, 256, 0, stream>>>(ei, eEx, denom, H, OUT);
        // emb = relu(out + b)   (in place in OUT)
        bias_relu<<<elemBlk, 256, 0, stream>>>(OUT, bias, nodeElems, D);
    };

    // layer 1: input x [N, F_IN]
    gat_layer(x, F_IN, W1, as1, ad1, b1);
    // layer 2: input emb1 (OUT); gemm writes H, OUT recycled as agg buffer
    gat_layer(OUT, D, W2, as2, ad2, b2);
    // classifier
    fc_softmax<<<nodeBlk, 256, 0, stream>>>(OUT, fcw, fcb, outp, N_NODES, D);
}
